// ComplexAttentivePooling_27977416966209
// MI455X (gfx1250) — compile-verified, matrix operand order checked by reading, not run on hardware
//
#include <hip/hip_runtime.h>

// Problem constants (from reference): B=16, S=8192, D=512, H=8, K=V=64.
#define B_      16
#define S_      8192
#define D_      512
#define H_      8
#define K_      64
#define V_      64
#define KSTEPS  16        // D / 32 (K-dim per v_wmma_f32_16x16x32_bf16)
#define EPS_    1e-12f

// Packed-weight layout (bf16), in per-stage consumption order:
//   per head h (8): 4 phase-1 stages of 32 KB  (qr_t & qi_t frags, s-major)
//                   4 phase-2 stages of 16 KB  (v_t frags, s-major)
//   -> 192 KB per head, 1.5 MB total.
#define HEAD_BYTES   196608u
#define P1_STAGE_B   32768u
#define P2_STAGE_B   16384u
#define PW_ELEMS     (96 * KSTEPS * 32 * 16)   // 786432 bf16 elements

typedef __attribute__((ext_vector_type(16))) __bf16 v16bf;
typedef __attribute__((ext_vector_type(8)))  float  v8f;

union FragCvt { uint4 u[2]; v16bf v; };

__device__ __forceinline__ unsigned pack_bf16(float lo, float hi) {
  unsigned short a = __builtin_bit_cast(unsigned short, (__bf16)lo);
  unsigned short b = __builtin_bit_cast(unsigned short, (__bf16)hi);
  return ((unsigned)b << 16) | (unsigned)a;
}

__device__ __forceinline__ v8f wmma_bf16(v16bf a, v16bf b, v8f c) {
  return __builtin_amdgcn_wmma_f32_16x16x32_bf16(false, a, false, b, (short)0, c,
                                                 false, false);
}

__device__ __forceinline__ v16bf ld_frag_lds(const char* p) {
  FragCvt f;
  f.u[0] = *(const uint4*)p;
  f.u[1] = *(const uint4*)(p + 16);
  return f.v;
}

// 16-lane butterfly reduction (stays within each half of the wave32).
__device__ __forceinline__ float red16(float v) {
  v += __shfl_xor(v, 1, 32);
  v += __shfl_xor(v, 2, 32);
  v += __shfl_xor(v, 4, 32);
  v += __shfl_xor(v, 8, 32);
  return v;
}

// CDNA5 async DMA: global -> LDS, 16B per lane, no VGPR staging.
// GVS mode: mem = SGPR64(base) + VGPR32(goff); LDS dest = LDS_BASE + VGPR(laddr).
__device__ __forceinline__ void async_g2l_b128(unsigned laddr, unsigned goff,
                                               const void* base) {
  asm volatile("global_load_async_to_lds_b128 %0, %1, %2"
               :: "v"(laddr), "v"(goff), "s"(base) : "memory");
}
__device__ __forceinline__ void wait_async0() {
  asm volatile("s_wait_asynccnt 0x0" ::: "memory");
}

// ---------------------------------------------------------------------------
// Kernel 1: pre-pack [Wq_r|Wq_i|Wv] (f32 [H,D,64]) into bf16 B-fragments in
// per-stage consumption order. Fragment element e of lane L maps to
// k = 32*s + e + (e>=8?8:0) + (L>=16?8:0), n = 16*t + (L&15)  (per §7.12.2).
// ---------------------------------------------------------------------------
__global__ void prepack_weights_kernel(const float* __restrict__ Wq_r,
                                       const float* __restrict__ Wq_i,
                                       const float* __restrict__ Wv,
                                       unsigned short* __restrict__ pw) {
  int E = blockIdx.x * blockDim.x + threadIdx.x;
  if (E >= PW_ELEMS) return;
  int h = E / 98304;           // 98304 bf16 elems per head
  int r = E % 98304;
  int t, s, lane, e, grp;      // grp: 0=qr, 1=qi, 2=v
  if (r < 65536) {             // phase-1 stages: [t][s][tile(2)][lane][16]
    t = r / 16384; int r1 = r % 16384;
    s = r1 / 1024; int r2 = r1 % 1024;
    grp = r2 / 512; int r3 = r2 % 512;
    lane = r3 / 16; e = r3 % 16;
  } else {                     // phase-2 stages: [t][s][lane][16]
    int r0 = r - 65536;
    t = r0 / 8192; int r1 = r0 % 8192;
    s = r1 / 512;  int r2 = r1 % 512;
    lane = r2 / 16; e = r2 % 16;
    grp = 2;
  }
  int kk = t * 16 + (lane & 15);                                    // out col
  int k  = s * 32 + e + (e >= 8 ? 8 : 0) + ((lane >= 16) ? 8 : 0);  // d index
  const float* W = (grp == 0) ? Wq_r : (grp == 1) ? Wq_i : Wv;
  float val = W[((size_t)h * D_ + k) * 64 + kk];
  pw[E] = __builtin_bit_cast(unsigned short, (__bf16)val);
}

// ---------------------------------------------------------------------------
// Kernel 2: complex L2-normalize keys (tiny).
// ---------------------------------------------------------------------------
__global__ void normalize_keys_kernel(const float* __restrict__ key_real,
                                      const float* __restrict__ key_imag,
                                      float* __restrict__ krn,
                                      float* __restrict__ kin) {
  int t = blockIdx.x * blockDim.x + threadIdx.x;
  if (t >= H_ * K_) return;
  int h = t >> 6;
  float ss = 0.f;
  for (int j = 0; j < K_; ++j) {
    float r = key_real[h * K_ + j], i = key_imag[h * K_ + j];
    ss += r * r + i * i;
  }
  float rn = rsqrtf(fmaxf(ss, EPS_));
  krn[t] = key_real[t] * rn;
  kin[t] = key_imag[t] * rn;
}

__global__ void zero_pooled_kernel(float* __restrict__ pooled) {
  int t = blockIdx.x * blockDim.x + threadIdx.x;
  if (t < B_ * H_ * V_) pooled[t] = 0.f;
}

// ---------------------------------------------------------------------------
// Kernel 3: fused projection + score + masked pooling.
// 256 threads = 8 waves; each wave owns 16 rows (A frags held in VGPRs, each
// reused 96x). B streams through a double-buffered 64 KB LDS pipeline filled
// by async global->LDS DMA and shared by all 8 waves (8x L2-traffic cut).
// Inner s-loops are software-pipelined: next B fragment's ds_loads stay in
// flight under the current WMMA (waits become dscnt<=N, not full drains).
// ---------------------------------------------------------------------------
__global__ __launch_bounds__(256) void fused_proj_pool_kernel(
    const float* __restrict__ x, const unsigned char* __restrict__ mask,
    const float* __restrict__ bq_r, const float* __restrict__ bq_i,
    const float* __restrict__ bv, const unsigned short* __restrict__ pw,
    const float* __restrict__ krn, const float* __restrict__ kin,
    float* __restrict__ pooled) {
  extern __shared__ char smem_dyn[];            // 2 x 32 KB B double-buffer
  const int tid  = threadIdx.x;
  const int lane = tid & 31;
  const int wave = tid >> 5;
  const int b    = blockIdx.x >> 6;             // 64 chunks of 128 rows / batch
  const int row0 = (blockIdx.x & 63) * 128 + wave * 16;
  const int m    = lane & 15;
  const int half = lane >> 4;
  const unsigned lds_base = (unsigned)(uintptr_t)smem_dyn;

  // ---- Stage A into registers: 16 rows x D, f32 -> bf16 fragments ----
  v16bf A[KSTEPS];
  {
    const float* xrow = x + ((size_t)b * S_ + (row0 + m)) * D_;
#pragma unroll
    for (int s = 0; s < KSTEPS; ++s) {
      const int d0 = s * 32 + half * 8;
      float4 f0 = *(const float4*)(xrow + d0);
      float4 f1 = *(const float4*)(xrow + d0 + 4);
      float4 f2 = *(const float4*)(xrow + d0 + 16);
      float4 f3 = *(const float4*)(xrow + d0 + 20);
      FragCvt fc;
      fc.u[0].x = pack_bf16(f0.x, f0.y); fc.u[0].y = pack_bf16(f0.z, f0.w);
      fc.u[0].z = pack_bf16(f1.x, f1.y); fc.u[0].w = pack_bf16(f1.z, f1.w);
      fc.u[1].x = pack_bf16(f2.x, f2.y); fc.u[1].y = pack_bf16(f2.z, f2.w);
      fc.u[1].z = pack_bf16(f3.x, f3.y); fc.u[1].w = pack_bf16(f3.z, f3.w);
      A[s] = fc.v;
    }
  }

  float mrow[8];
#pragma unroll
  for (int r = 0; r < 8; ++r)
    mrow[r] = mask[(size_t)b * S_ + row0 + r + half * 8] ? 1.0f : 0.0f;

  const v8f zacc = {0.f, 0.f, 0.f, 0.f, 0.f, 0.f, 0.f, 0.f};

  auto stage_goff = [](int st) -> unsigned {
    int h = st >> 3, sub = st & 7;
    return (unsigned)h * HEAD_BYTES +
           (sub < 4 ? (unsigned)sub * P1_STAGE_B
                    : 4u * P1_STAGE_B + (unsigned)(sub - 4) * P2_STAGE_B);
  };
  auto stage_bytes = [](int st) -> unsigned {
    return ((st & 7) < 4) ? P1_STAGE_B : P2_STAGE_B;
  };
  auto issue_stage = [&](int st) {
    unsigned base = stage_goff(st);
    unsigned lbuf = lds_base + ((st & 1) ? P1_STAGE_B : 0u);
    unsigned nb   = stage_bytes(st);
    for (unsigned off = (unsigned)tid * 16; off < nb; off += 4096)
      async_g2l_b128(lbuf + off, base + off, pw);
  };

  issue_stage(0);

  float ss[8], dt[8], score[8];
  for (int st = 0; st < 64; ++st) {
    wait_async0();        // this wave's share of stage st landed in LDS
    __syncthreads();      // everyone's share landed; prev compute done
    if (st + 1 < 64) issue_stage(st + 1);   // DMA overlaps compute below

    const char* bbuf = smem_dyn + ((st & 1) ? P1_STAGE_B : 0u);
    const int h = st >> 3, sub = st & 7;
    if (sub < 4) {
      // ---- phase 1: one (qr_t, qi_t) tile pair; fold into ss/dt ----
      if (sub == 0) {
#pragma unroll
        for (int r = 0; r < 8; ++r) { ss[r] = 0.f; dt[r] = 0.f; }
      }
      v8f aq = zacc, ai = zacc;
      // Software pipeline: keep the next fragment pair's ds_loads in flight.
      v16bf cq = ld_frag_lds(bbuf + (0 * 32 + lane) * 32);
      v16bf ci = ld_frag_lds(bbuf + (1 * 32 + lane) * 32);
#pragma unroll
      for (int s = 0; s < KSTEPS; ++s) {
        v16bf nq = cq, ni = ci;
        if (s + 1 < KSTEPS) {
          nq = ld_frag_lds(bbuf + (((s + 1) * 2 + 0) * 32 + lane) * 32);
          ni = ld_frag_lds(bbuf + (((s + 1) * 2 + 1) * 32 + lane) * 32);
        }
        aq = wmma_bf16(A[s], cq, aq);
        ai = wmma_bf16(A[s], ci, ai);
        cq = nq; ci = ni;
      }
      const int col = h * K_ + sub * 16 + (lane & 15);
      const float krv = krn[col], kiv = kin[col];
      const float br = bq_r[col], bic = bq_i[col];
#pragma unroll
      for (int r = 0; r < 8; ++r) {
        float qr = aq[r] + br;
        float qi = ai[r] + bic;
        ss[r] += qr * qr + qi * qi;
        dt[r] += qr * krv - qi * kiv;
      }
      if (sub == 3) {
#pragma unroll
        for (int r = 0; r < 8; ++r)
          score[r] = mrow[r] * red16(dt[r]) * rsqrtf(fmaxf(red16(ss[r]), EPS_));
      }
    } else {
      // ---- phase 2: one v_t tile; weight by score, pool over rows ----
      const int t = sub - 4;
      v8f av = zacc;
      v16bf cv = ld_frag_lds(bbuf + (0 * 32 + lane) * 32);
#pragma unroll
      for (int s = 0; s < KSTEPS; ++s) {
        v16bf nv = cv;
        if (s + 1 < KSTEPS)
          nv = ld_frag_lds(bbuf + ((s + 1) * 32 + lane) * 32);
        av = wmma_bf16(A[s], cv, av);
        cv = nv;
      }
      const int col = h * V_ + t * 16 + (lane & 15);
      const float bvv = bv[col];
      float ps = 0.f;
#pragma unroll
      for (int r = 0; r < 8; ++r) ps += score[r] * (av[r] + bvv);
      // Each half contributes its 8 rows; both halves hit the same column.
      atomicAdd(&pooled[(size_t)b * (H_ * V_) + col], ps);
    }
  }
}

// ---------------------------------------------------------------------------
// Kernel 4: final feedforward  out[b,d] = bf[d] + sum_n pooled[b,n]*Wf[n,d].
// ---------------------------------------------------------------------------
__global__ void final_ff_kernel(const float* __restrict__ pooled,
                                const float* __restrict__ Wf,
                                const float* __restrict__ bf,
                                float* __restrict__ out) {
  int d = blockIdx.x * blockDim.x + threadIdx.x;
  int b = blockIdx.y;
  if (d >= D_) return;
  float acc = bf[d];
  for (int n = 0; n < H_ * V_; ++n)
    acc = fmaf(pooled[(size_t)b * (H_ * V_) + n], Wf[(size_t)n * D_ + d], acc);
  out[(size_t)b * D_ + d] = acc;
}

// ---------------------------------------------------------------------------
extern "C" void kernel_launch(void* const* d_in, const int* in_sizes, int n_in,
                              void* d_out, int out_size, void* d_ws,
                              size_t ws_size, hipStream_t stream) {
  const float*         x        = (const float*)d_in[0];
  const unsigned char* mask     = (const unsigned char*)d_in[1];
  const float*         Wq_r     = (const float*)d_in[2];
  const float*         bq_r     = (const float*)d_in[3];
  const float*         Wq_i     = (const float*)d_in[4];
  const float*         bq_i     = (const float*)d_in[5];
  const float*         Wv       = (const float*)d_in[6];
  const float*         bv       = (const float*)d_in[7];
  const float*         key_real = (const float*)d_in[8];
  const float*         key_imag = (const float*)d_in[9];
  const float*         Wf       = (const float*)d_in[10];
  const float*         bf       = (const float*)d_in[11];
  float*               out      = (float*)d_out;

  char* ws = (char*)d_ws;
  unsigned short* pw = (unsigned short*)ws;                  // 1,572,864 B
  float* pooled = (float*)(ws + (size_t)PW_ELEMS * 2);       // 32 KB
  float* krn    = pooled + B_ * H_ * V_;
  float* kin    = krn + H_ * K_;

  prepack_weights_kernel<<<(PW_ELEMS + 255) / 256, 256, 0, stream>>>(Wq_r, Wq_i,
                                                                     Wv, pw);
  normalize_keys_kernel<<<2, 256, 0, stream>>>(key_real, key_imag, krn, kin);
  zero_pooled_kernel<<<(B_ * H_ * V_ + 255) / 256, 256, 0, stream>>>(pooled);

  // 1024 blocks x 256 threads; each block = 128 rows; 64 KB dynamic LDS.
  fused_proj_pool_kernel<<<1024, 256, 2 * P1_STAGE_B, stream>>>(
      x, mask, bq_r, bq_i, bv, pw, krn, kin, pooled);

  final_ff_kernel<<<dim3(2, B_), 256, 0, stream>>>(pooled, Wf, bf, out);
}
